// BinLinearEval_34359738368299
// MI455X (gfx1250) — matrix operand order repaired and checked by reading, not compiled, not run on hardware
//
#include <hip/hip_runtime.h>
#include <hip/hip_bf16.h>

typedef __attribute__((ext_vector_type(16))) __bf16 v16bf;
typedef __attribute__((ext_vector_type(8)))  float  v8f;
typedef __attribute__((ext_vector_type(4)))  float  v4f;
typedef __attribute__((ext_vector_type(4)))  int    v4i;

#if __has_builtin(__builtin_amdgcn_global_load_async_to_lds_b128) && \
    __has_builtin(__builtin_amdgcn_s_wait_asynccnt)
#define USE_ASYNC 1
typedef __attribute__((address_space(1))) v4i* gv4i_t;   // global int4*
typedef __attribute__((address_space(3))) v4i* lv4i_t;   // LDS int4*
#else
#define USE_ASYNC 0
#endif

// Round-to-nearest-even f32 -> bf16, packed pair into one dword.
__device__ __forceinline__ unsigned pk_bf16(float a, float b) {
    union { float f; unsigned u; } x, y;
    x.f = a; y.f = b;
    unsigned ua = x.u + 0x7FFFu + ((x.u >> 16) & 1u);
    unsigned ub = y.u + 0x7FFFu + ((y.u >> 16) & 1u);
    return (ua >> 16) | (ub & 0xFFFF0000u);
}

// ---------------------------------------------------------------------------
// Prologue: ternary weight [256,1024] f32 -> bf16 row-major in ws (512 KB).
// ---------------------------------------------------------------------------
__global__ __launch_bounds__(256) void cvt_weight_bf16(
    const float* __restrict__ w, unsigned* __restrict__ wb)
{
    int i = blockIdx.x * 256 + threadIdx.x;     // 512 * 256 = 131072 dwords
    int j = i * 2;
    wb[i] = pk_bf16(w[j], w[j + 1]);
}

// ---------------------------------------------------------------------------
// Main GEMM: M=65536, N=256, K=1024. 512 WGs x 256 threads (8 waves).
// Wave owns 16 rows x all 256 cols (16 acc tiles). K chunked by 32.
// ---------------------------------------------------------------------------
#define B_PITCH 40   // shorts per LDS row (80B: 16B-aligned, conflict-free b128)

struct AF { v4f a0, a1, a2, a3; };

__device__ __forceinline__ void load_A(const float* p, AF& A) {
    // lane<16: K 0..15; lane>=16: K 16..31 (khalf folded in p)
    A.a0 = __builtin_nontemporal_load((const v4f*)(p));
    A.a1 = __builtin_nontemporal_load((const v4f*)(p + 4));
    A.a2 = __builtin_nontemporal_load((const v4f*)(p + 8));
    A.a3 = __builtin_nontemporal_load((const v4f*)(p + 12));
}

__device__ __forceinline__ void compute_step(
    const unsigned short* __restrict__ buf,   // LDS W slice [256][B_PITCH]
    int nlane, int khalf, const AF& Ar, v8f acc[16])
{
    union { v16bf v; unsigned u[8]; } A;
    A.u[0] = pk_bf16(Ar.a0.x, Ar.a0.y); A.u[1] = pk_bf16(Ar.a0.z, Ar.a0.w);
    A.u[2] = pk_bf16(Ar.a1.x, Ar.a1.y); A.u[3] = pk_bf16(Ar.a1.z, Ar.a1.w);
    A.u[4] = pk_bf16(Ar.a2.x, Ar.a2.y); A.u[5] = pk_bf16(Ar.a2.z, Ar.a2.w);
    A.u[6] = pk_bf16(Ar.a3.x, Ar.a3.y); A.u[7] = pk_bf16(Ar.a3.z, Ar.a3.w);

    // B frag: lane<16 -> N=nlane, K 0..15; lane>=16 -> N=nlane, K 16..31.
    const unsigned short* base = buf + nlane * B_PITCH + khalf * 16;
    union BU { v16bf v; uint4 q[2]; };
    BU B[2];                                  // ping-pong: overlap ds w/ wmma
    B[0].q[0] = ((const uint4*)base)[0];
    B[0].q[1] = ((const uint4*)base)[1];
#pragma unroll
    for (int nt = 0; nt < 16; ++nt) {
        if (nt < 15) {
            const uint4* bs = (const uint4*)(base + (nt + 1) * 16 * B_PITCH);
            B[(nt + 1) & 1].q[0] = bs[0];
            B[(nt + 1) & 1].q[1] = bs[1];
        }
        acc[nt] = __builtin_amdgcn_wmma_f32_16x16x32_bf16(
            false, A.v, false, B[nt & 1].v, (short)0, acc[nt], false, false);
    }
}

__global__ __launch_bounds__(256) void binlinear_wmma(
    const float* __restrict__ x,
    const unsigned short* __restrict__ wb,   // bf16 bits, [256][1024]
    const float* __restrict__ bias,
    const float* __restrict__ sgn,
    float* __restrict__ out)
{
    __shared__ unsigned short ldsB[2][256 * B_PITCH];   // 2 x 20 KB

    const int t     = threadIdx.x;
    const int lane  = t & 31;
    const int wv    = t >> 5;
    const int nlane = lane & 15;
    const int khalf = lane >> 4;

    const long m0 = (long)blockIdx.x * 128 + wv * 16;
    const float* xrow = x + (m0 + nlane) * 1024 + khalf * 16;

    v8f zero = {0.f, 0.f, 0.f, 0.f, 0.f, 0.f, 0.f, 0.f};
    v8f acc[16];
#pragma unroll
    for (int i = 0; i < 16; ++i) acc[i] = zero;

#if USE_ASYNC
    // Async W staging: each thread DMA-copies 64B of its W row per K-step.
    // Builtin order (from clang's type check): (global int4*, lds int4*,
    // imm offset applied to BOTH sides, imm cpol).
    gv4i_t gW = (gv4i_t)(wb + (size_t)t * 1024);
    lv4i_t lB0 = (lv4i_t)(&ldsB[0][t * B_PITCH]);
    lv4i_t lB1 = (lv4i_t)(&ldsB[1][t * B_PITCH]);

    // prologue: stage slice 0, load A(0)
    __builtin_amdgcn_global_load_async_to_lds_b128(gW, lB0,  0, 0);
    __builtin_amdgcn_global_load_async_to_lds_b128(gW, lB0, 16, 0);
    __builtin_amdgcn_global_load_async_to_lds_b128(gW, lB0, 32, 0);
    __builtin_amdgcn_global_load_async_to_lds_b128(gW, lB0, 48, 0);
    AF Acur; load_A(xrow, Acur);
    __builtin_amdgcn_s_wait_asynccnt(0);
    __syncthreads();

#pragma unroll 2
    for (int ks = 0; ks < 32; ++ks) {
        if (ks < 31) {   // issue next W slice into other buffer (async)
            gv4i_t g = gW + (ks + 1) * 4;              // +64B per K-step
            lv4i_t l = ((ks + 1) & 1) ? lB1 : lB0;
            __builtin_amdgcn_global_load_async_to_lds_b128(g, l,  0, 0);
            __builtin_amdgcn_global_load_async_to_lds_b128(g, l, 16, 0);
            __builtin_amdgcn_global_load_async_to_lds_b128(g, l, 32, 0);
            __builtin_amdgcn_global_load_async_to_lds_b128(g, l, 48, 0);
        }
        AF Anext;
        if (ks < 31) load_A(xrow + (ks + 1) * 32, Anext);   // prefetch x
        compute_step(&ldsB[ks & 1][0], nlane, khalf, Acur, acc);
        if (ks < 31) {
            __builtin_amdgcn_s_wait_asynccnt(0);
            __syncthreads();
            Acur = Anext;
        }
    }
#else
    // Sync fallback: stage via registers, double-buffered, 1 barrier/step.
    const uint4* wsrc = (const uint4*)(wb + (size_t)t * 1024);
    {
        uint4 r0 = wsrc[0], r1 = wsrc[1], r2 = wsrc[2], r3 = wsrc[3];
        uint4* dst = (uint4*)(&ldsB[0][t * B_PITCH]);
        dst[0] = r0; dst[1] = r1; dst[2] = r2; dst[3] = r3;
    }
    AF Acur; load_A(xrow, Acur);
    __syncthreads();

#pragma unroll 2
    for (int ks = 0; ks < 32; ++ks) {
        uint4 r0, r1, r2, r3;
        if (ks < 31) {   // prefetch next W slice into regs
            const uint4* s = wsrc + (ks + 1) * 4;
            r0 = s[0]; r1 = s[1]; r2 = s[2]; r3 = s[3];
        }
        AF Anext;
        if (ks < 31) load_A(xrow + (ks + 1) * 32, Anext);   // prefetch x
        compute_step(&ldsB[ks & 1][0], nlane, khalf, Acur, acc);
        if (ks < 31) {
            uint4* dst = (uint4*)(&ldsB[(ks + 1) & 1][t * B_PITCH]);
            dst[0] = r0; dst[1] = r1; dst[2] = r2; dst[3] = r3;
            __syncthreads();
            Acur = Anext;
        }
    }
#endif

    // ---- epilogue: y = rint(acc + bias); out = (y*sign >= 0) ? 1 : 0
    // acc tile: VGPR g -> row m0 + g + khalf*8, col nt*16 + nlane.
#pragma unroll
    for (int nt = 0; nt < 16; ++nt) {
        const int col = nt * 16 + nlane;
        const float bv = bias[col];
        const float sv = sgn[col];
        const long rbase = m0 + khalf * 8;
#pragma unroll
        for (int g = 0; g < 8; ++g) {
            float y = rintf(acc[nt][g] + bv);
            float r = (y * sv >= 0.0f) ? 1.0f : 0.0f;
            __builtin_nontemporal_store(r, &out[(rbase + g) * 256 + col]);
        }
    }
}

extern "C" void kernel_launch(void* const* d_in, const int* in_sizes, int n_in,
                              void* d_out, int out_size, void* d_ws, size_t ws_size,
                              hipStream_t stream) {
    const float* x    = (const float*)d_in[0];   // [65536,1024]
    const float* w    = (const float*)d_in[1];   // [256,1024] ternary
    const float* bias = (const float*)d_in[2];   // [256]
    const float* sgn  = (const float*)d_in[3];   // [256]
    unsigned short* wb = (unsigned short*)d_ws;  // 512 KB bf16 weights
    float* out = (float*)d_out;

    cvt_weight_bf16<<<512, 256, 0, stream>>>(w, (unsigned*)wb);
    binlinear_wmma<<<512, 256, 0, stream>>>(x, wb, bias, sgn, out);

    (void)in_sizes; (void)n_in; (void)out_size; (void)ws_size;
}
